// GCNN_56796647522693
// MI455X (gfx1250) — matrix operand order repaired and checked by reading, not compiled, not run on hardware
//
#include <hip/hip_runtime.h>

typedef __attribute__((ext_vector_type(16))) _Float16 v16h;
typedef __attribute__((ext_vector_type(8)))  float    v8f;

#define BNKS 1280
#define LTOK 64
#define DIN  512
#define DOUT 256

#define SC 260        // f32 row stride for P/S (DOUT + 4 pad, keeps 16B align, breaks conflicts)
#define ALANE 18      // dwords per lane per A fragment slot (8 hi + 8 lo + 2 pad);
                      // gcd(18,64)=2 -> lane bank offsets 18*l mod 64 all distinct for l<32

// A fragment store: [rt 4][ks 16][lane 32][ALANE dwords]
#define A_DWORDS (4 * 16 * 32 * ALANE)   // 36864 dwords = 147456 B

struct SMem {
  union {
    unsigned int A[A_DWORDS];                              // live during GEMM (147456 B)
    struct { float P[LTOK * SC]; float S[LTOK * SC]; } r;  // live after GEMM (133120 B)
  } u;
  float ginTot[LTOK];
  float gsTot [LTOK];
  int   head[LTOK];
  int   lab [LTOK];
  float win[LTOK];
  float wsl[LTOK];
  float mk [LTOK];
};

union F16Frag { v16h v; unsigned int u32[8]; uint4 u4[2]; };
union PK2     { _Float16 h[2]; unsigned int u; };

__device__ __forceinline__ float sigmoidf_(float x) { return 1.0f / (1.0f + __expf(-x)); }

// ---------------------------------------------------------------------------
// Pre-kernel: split W_in/W_self into hi/lo f16 and store in exact B-fragment
// layout: ws dword index = ((mat*256 + ct*16 + ks)*32 + lane)*16, [0..7]=hi, [8..15]=lo.
// Runs once per launch; total ws = 2*16*16*32*16 dwords = 1 MiB.
// ---------------------------------------------------------------------------
__global__ void wswizzle(const float* __restrict__ W_in,
                         const float* __restrict__ W_self,
                         unsigned int* __restrict__ ws)
{
  const int t    = threadIdx.x;
  const int wave = t >> 5;
  const int lane = t & 31;
  const int half = lane >> 4;
  const int n    = lane & 15;
  const int gf   = blockIdx.x * 8 + wave;     // [0, 512)
  const int mat  = gf >> 8;
  const int ct   = (gf >> 4) & 15;
  const int ks   = gf & 15;
  const float* Wp = mat ? W_self : W_in;
  const float* wb = Wp + (size_t)(ks * 32 + 16 * half) * DOUT + ct * 16 + n;
  unsigned int* dst = ws + ((size_t)gf * 32 + lane) * 16;
#pragma unroll
  for (int v = 0; v < 8; ++v) {
    float f0 = wb[v * 2 * DOUT];
    float f1 = wb[(v * 2 + 1) * DOUT];
    _Float16 h0 = (_Float16)f0, h1 = (_Float16)f1;
    PK2 ph, pl;
    ph.h[0] = h0;
    ph.h[1] = h1;
    pl.h[0] = (_Float16)(f0 - (float)h0);
    pl.h[1] = (_Float16)(f1 - (float)h1);
    dst[v]     = ph.u;
    dst[8 + v] = pl.u;
  }
}

// ---------------------------------------------------------------------------
// Fused GCNN kernel: one workgroup (8 wave32) per bank of 64 tokens.
// ---------------------------------------------------------------------------
__launch_bounds__(256, 1)
__global__ void gcnn_fused(const float* __restrict__ rep,
                           const float* __restrict__ adj_mask_in,
                           const float* __restrict__ adj_mask_loop,
                           const float* __restrict__ mask,
                           const float* __restrict__ b_in,
                           const float* __restrict__ W_gate_in,
                           const float* __restrict__ b_gate_in,
                           const float* __restrict__ W_gate_self,
                           const int*   __restrict__ adj_arc_in,
                           const int*   __restrict__ adj_lab_in,
                           const unsigned int* __restrict__ ws,
                           float* __restrict__ out)
{
  __shared__ SMem sm;
  const int t    = threadIdx.x;
  const int bank = blockIdx.x;
  const int wave = t >> 5;
  const int lane = t & 31;
  const int half = lane >> 4;
  const int n    = lane & 15;

  // ---- Phase 1: rep block -> split hi/lo f16, pre-swizzled A fragments in LDS
  const float4* rep4 = (const float4*)(rep + (size_t)bank * LTOK * DIN);
  for (int i = 0; i < 32; ++i) {
    int e = t + i * 256;               // float4 index in [0, 8192)
    float4 g = rep4[e];
    int fidx = e * 4;
    int r   = fidx >> 9;               // token row
    int kb4 = fidx & (DIN - 1);        // k base (multiple of 4)
    int rt = r >> 4, nn = r & 15;
    int ks = kb4 >> 5;
    int kk = kb4 & 31;
    int hf = kk >> 4;                  // lane half holding this K range
    int v  = (kk & 15) >> 1;           // first dword slot (covers v, v+1)
    unsigned int* bd = &sm.u.A[(((rt * 16 + ks) * 32) + (hf * 16 + nn)) * ALANE + v];
    _Float16 h0 = (_Float16)g.x, h1 = (_Float16)g.y, h2 = (_Float16)g.z, h3 = (_Float16)g.w;
    PK2 p0, p1, q0, q1;
    p0.h[0] = h0; p0.h[1] = h1;
    p1.h[0] = h2; p1.h[1] = h3;
    q0.h[0] = (_Float16)(g.x - (float)h0); q0.h[1] = (_Float16)(g.y - (float)h1);
    q1.h[0] = (_Float16)(g.z - (float)h2); q1.h[1] = (_Float16)(g.w - (float)h3);
    bd[0] = p0.u;
    bd[1] = p1.u;
    bd[8] = q0.u;
    bd[9] = q1.u;
  }

  // ---- Phase 2: gate GEMVs from global rep (L2-hot), wave-parallel ---------
  {
    const float* repB = rep + (size_t)bank * LTOK * DIN;
    for (int rr = 0; rr < 8; ++rr) {
      int r = wave * 8 + rr;
      const float* row = repB + (size_t)r * DIN;
      float gin = 0.f, gs = 0.f;
#pragma unroll
      for (int j = 0; j < 16; ++j) {
        float a = row[j * 32 + lane];
        gin = fmaf(a, W_gate_in[j * 32 + lane], gin);
        gs  = fmaf(a, W_gate_self[j * 32 + lane], gs);
      }
#pragma unroll
      for (int m = 16; m >= 1; m >>= 1) {
        gin += __shfl_xor(gin, m, 32);
        gs  += __shfl_xor(gs, m, 32);
      }
      if (lane == 0) { sm.ginTot[r] = gin; sm.gsTot[r] = gs; }
    }
  }
  __syncthreads();

  if (t < LTOK) {
    int   h    = adj_arc_in[(bank * LTOK + t) * 2 + 1];
    int   lb   = adj_lab_in[bank * LTOK + t];
    float msin = adj_mask_in  [bank * LTOK + t];
    float msl  = adj_mask_loop[bank * LTOK + t];
    sm.head[t] = h;
    sm.lab [t] = lb;
    sm.win [t] = sigmoidf_(sm.ginTot[h] + b_gate_in[lb]) * msin * msin;
    sm.wsl [t] = sigmoidf_(sm.gsTot[t]) * msl * msl;
    sm.mk  [t] = mask[bank * LTOK + t];
  }
  __syncthreads();

  // ---- Phase 3: dual GEMM via split-f16 WMMA --------------------------------
  // waves 0-3: P = rep @ W_in ; waves 4-7: S = rep @ W_self
  // each wave: 4 column tiles, 4 row tiles, K = 512 in 16 steps of 32
  const int mat = wave >> 2;
  const unsigned int* wsw = ws + (size_t)mat * (16 * 16 * 32 * 16);

  v8f acc[4][4];
  const v8f vzero = {0.f, 0.f, 0.f, 0.f, 0.f, 0.f, 0.f, 0.f};
#pragma unroll
  for (int u = 0; u < 4; ++u)
#pragma unroll
    for (int rt = 0; rt < 4; ++rt)
      acc[u][rt] = vzero;

  for (int ks = 0; ks < 16; ++ks) {
    F16Frag Ah[4], Al[4];
#pragma unroll
    for (int rt = 0; rt < 4; ++rt) {
      const unsigned int* ab = &sm.u.A[((rt * 16 + ks) * 32 + lane) * ALANE];
#pragma unroll
      for (int v = 0; v < 8; v += 2) {
        uint2 qh = *(const uint2*)(ab + v);
        uint2 ql = *(const uint2*)(ab + 8 + v);
        Ah[rt].u32[v]     = qh.x;
        Ah[rt].u32[v + 1] = qh.y;
        Al[rt].u32[v]     = ql.x;
        Al[rt].u32[v + 1] = ql.y;
      }
    }
#pragma unroll
    for (int u = 0; u < 4; ++u) {
      const int ct = (wave & 3) * 4 + u;
      const uint4* wb = (const uint4*)(wsw + ((size_t)(ct * 16 + ks) * 32 + lane) * 16);
      F16Frag Bh, Bl;
      Bh.u4[0] = wb[0];
      Bh.u4[1] = wb[1];
      Bl.u4[0] = wb[2];
      Bl.u4[1] = wb[3];
#pragma unroll
      for (int rt = 0; rt < 4; ++rt) {
        acc[u][rt] = __builtin_amdgcn_wmma_f32_16x16x32_f16(false, Ah[rt].v, false, Bh.v,
                                                            (short)0, acc[u][rt], false, false);
        acc[u][rt] = __builtin_amdgcn_wmma_f32_16x16x32_f16(false, Ah[rt].v, false, Bl.v,
                                                            (short)0, acc[u][rt], false, false);
        acc[u][rt] = __builtin_amdgcn_wmma_f32_16x16x32_f16(false, Al[rt].v, false, Bh.v,
                                                            (short)0, acc[u][rt], false, false);
      }
    }
  }

  __syncthreads();   // A region dead everywhere; safe to overwrite union with P/S

  {
    float* dst = mat ? sm.u.r.S : sm.u.r.P;
#pragma unroll
    for (int u = 0; u < 4; ++u) {
      const int c0 = ((wave & 3) * 4 + u) * 16;
#pragma unroll
      for (int rt = 0; rt < 4; ++rt) {
#pragma unroll
        for (int i = 0; i < 8; ++i) {
          int row = rt * 16 + 8 * half + i;   // C/D layout: VGPR i -> M=i (lo half) / i+8 (hi half)
          dst[row * SC + c0 + n] = acc[u][rt][i];
        }
      }
    }
  }
  __syncthreads();

  // ---- Phase 4: gather + gated combine + ReLU + mask ------------------------
  float* outB = out + (size_t)bank * LTOK * DOUT;
  for (int i = 0; i < 16; ++i) {
    int idx = t + i * 256;                   // [0, 4096) float4 elements
    int r   = idx >> 6;
    int c   = (idx & 63) * 4;
    int h   = sm.head[r];
    int lb  = sm.lab[r];
    float wi = sm.win[r], wsf = sm.wsl[r], m = sm.mk[r];
    const float4 p  = *(const float4*)&sm.u.r.P[h * SC + c];
    const float4 s  = *(const float4*)&sm.u.r.S[r * SC + c];
    const float4 bi = *(const float4*)&b_in[lb * DOUT + c];
    float4 o;
    o.x = fmaxf(0.f, (p.x + bi.x) * wi + s.x * wsf) * m;
    o.y = fmaxf(0.f, (p.y + bi.y) * wi + s.y * wsf) * m;
    o.z = fmaxf(0.f, (p.z + bi.z) * wi + s.z * wsf) * m;
    o.w = fmaxf(0.f, (p.w + bi.w) * wi + s.w * wsf) * m;
    *(float4*)&outB[r * DOUT + c] = o;
  }
}

extern "C" void kernel_launch(void* const* d_in, const int* in_sizes, int n_in,
                              void* d_out, int out_size, void* d_ws, size_t ws_size,
                              hipStream_t stream) {
  (void)in_sizes; (void)n_in; (void)out_size; (void)ws_size; // requires ws_size >= 1 MiB
  unsigned int* ws = (unsigned int*)d_ws;

  // one-time (per launch) weight split + swizzle into d_ws
  wswizzle<<<64, 256, 0, stream>>>((const float*)d_in[4],   // W_in
                                   (const float*)d_in[8],   // W_self
                                   ws);

  gcnn_fused<<<BNKS, 256, 0, stream>>>(
      (const float*)d_in[0],   // rep
      (const float*)d_in[1],   // adj_mask_in
      (const float*)d_in[2],   // adj_mask_loop
      (const float*)d_in[3],   // mask
      (const float*)d_in[5],   // b_in
      (const float*)d_in[6],   // W_gate_in
      (const float*)d_in[7],   // b_gate_in
      (const float*)d_in[9],   // W_gate_self
      (const int*)  d_in[10],  // adj_arc_in
      (const int*)  d_in[11],  // adj_lab_in
      ws,
      (float*)d_out);
}
